// RBM_45397804319408
// MI455X (gfx1250) — matrix operand order repaired and checked by reading, not compiled, not run on hardware
//
#include <hip/hip_runtime.h>
#include <hip/hip_bf16.h>
#include <stdint.h>

// ---------------------------------------------------------------------------
// RBM CD-k gradient on MI455X (gfx1250): bf16 WMMA everywhere on the hot path.
//   B=16384 samples, V=1024 visible, H=256 hidden, k=8 Gibbs steps.
// Register-blocked 16x64 strips (4 WMMA chains sharing one A fragment) with
// explicit software pipelining: fragments for step k+32 are loaded before the
// WMMA burst on step k, so waits on the matrix operands are partial and VMEM
// overlaps the matrix pipe.
// ---------------------------------------------------------------------------

#define BSZ 16384
#define VSZ 1024
#define HSZ 256
#define KGIBBS 8

typedef __bf16 bf16;
typedef bf16  v16bf __attribute__((ext_vector_type(16)));
typedef bf16  v8bf  __attribute__((ext_vector_type(8)));
typedef float v8f   __attribute__((ext_vector_type(8)));

// ---- stateless hash RNG: uniform in [0,1) -------------------------------
__device__ __forceinline__ float rnd01(uint32_t x, uint32_t seed) {
    x ^= seed * 0x9E3779B9u;
    x ^= x >> 16; x *= 0x7FEB352Du;
    x ^= x >> 15; x *= 0x846CA68Bu;
    x ^= x >> 16;
    return (float)(x >> 8) * (1.0f / 16777216.0f);
}

// fast sigmoid: single v_rcp_f32 instead of the IEEE divide expansion
__device__ __forceinline__ float sigmoidf(float x) {
    return __builtin_amdgcn_rcpf(1.0f + __expf(-x));
}

// Build a 16-wide bf16 A fragment from two contiguous 8-element halves.
__device__ __forceinline__ v16bf make_a(const bf16* p) {
    v8bf lo = *(const v8bf*)(p);        // K = koff .. koff+7
    v8bf hi = *(const v8bf*)(p + 8);    // K = koff+8 .. koff+15
    v16bf a;
#pragma unroll
    for (int i = 0; i < 8; ++i) { a[i] = lo[i]; a[i + 8] = hi[i]; }
    return a;
}

// ---------------------------------------------------------------------------
// Kernel: convert W (f32 [H,V]) -> Wbf [H,V] and WTbf [V,H] (bf16)
// ---------------------------------------------------------------------------
__global__ void convert_w_kernel(const float* __restrict__ W,
                                 bf16* __restrict__ Wbf,
                                 bf16* __restrict__ WTbf) {
    int idx = blockIdx.x * blockDim.x + threadIdx.x;
    if (idx >= HSZ * VSZ) return;
    int h = idx / VSZ, v = idx % VSZ;
    float w = W[idx];
    Wbf[idx]          = (bf16)w;
    WTbf[v * HSZ + h] = (bf16)w;
}

// ---------------------------------------------------------------------------
// Kernel: convert batch (f32 [B,V]) -> batch_bf and initial vbuf (bf16)
// ---------------------------------------------------------------------------
__global__ void convert_batch_kernel(const float* __restrict__ batch,
                                     bf16* __restrict__ batch_bf,
                                     bf16* __restrict__ vbuf) {
    size_t idx = (size_t)blockIdx.x * blockDim.x + threadIdx.x;
    if (idx >= (size_t)BSZ * VSZ) return;
    bf16 b = (bf16)batch[idx];
    batch_bf[idx] = b;
    vbuf[idx]     = b;
}

__global__ void zero_f32_kernel(float* __restrict__ p, int n) {
    int idx = blockIdx.x * blockDim.x + threadIdx.x;
    if (idx < n) p[idx] = 0.0f;
}

// ---------------------------------------------------------------------------
// Hidden pass: pre[b,h] = sum_v vin[b,v]*W[h,v] + hb[h]   (WMMA bf16->f32)
//   mode 0: sample h  -> hout (bf16 {0,1})
//   mode 1: ph        -> phout (f32)
//   mode 2: ph -> phout, and atomicAdd softplus(pre) into logp[b]
// grid = (B/16, H/64), block = 32: one wave computes a 16x64 strip.
// ---------------------------------------------------------------------------
__global__ __launch_bounds__(32)
void hidden_wmma_kernel(const bf16* __restrict__ vin,
                        const bf16* __restrict__ Wbf,
                        const float* __restrict__ hb,
                        bf16* __restrict__ hout,
                        float* __restrict__ phout,
                        float* __restrict__ logp,
                        int mode, uint32_t seed) {
    const int b0   = blockIdx.x * 16;
    const int h0   = blockIdx.y * 64;
    const int lane = threadIdx.x;
    const int mrow = lane & 15;           // A-tile row
    const int koff = (lane >> 4) * 16;    // A-tile K offset
    const int ncol = lane & 15;           // B-tile column
    const int khalf = (lane >> 4) * 16;   // B-tile K offset

    v8f c[4] = {};
    const bf16* arow = vin + (size_t)(b0 + mrow) * VSZ + koff;
    const bf16* brow[4];
#pragma unroll
    for (int j = 0; j < 4; ++j)
        brow[j] = Wbf + (size_t)(h0 + 16 * j + ncol) * VSZ + khalf;

    // --- software-pipelined K loop: preload, then load(k+32) / compute(k) ---
    v16bf aC  = make_a(arow);
    v16bf bC0 = *(const v16bf*)(brow[0]);
    v16bf bC1 = *(const v16bf*)(brow[1]);
    v16bf bC2 = *(const v16bf*)(brow[2]);
    v16bf bC3 = *(const v16bf*)(brow[3]);

#pragma unroll 2
    for (int k0 = 0; k0 < VSZ - 32; k0 += 32) {
        const int k1 = k0 + 32;
        if ((k1 & 32) == 0)                       // one prefetch per cacheline
            __builtin_prefetch(arow + k1 + 128, 0, 1);
        v16bf aN  = make_a(arow + k1);
        v16bf bN0 = *(const v16bf*)(brow[0] + k1);
        v16bf bN1 = *(const v16bf*)(brow[1] + k1);
        v16bf bN2 = *(const v16bf*)(brow[2] + k1);
        v16bf bN3 = *(const v16bf*)(brow[3] + k1);
        c[0] = __builtin_amdgcn_wmma_f32_16x16x32_bf16(false, aC, false, bC0,
                                                       (short)0, c[0], false, false);
        c[1] = __builtin_amdgcn_wmma_f32_16x16x32_bf16(false, aC, false, bC1,
                                                       (short)0, c[1], false, false);
        c[2] = __builtin_amdgcn_wmma_f32_16x16x32_bf16(false, aC, false, bC2,
                                                       (short)0, c[2], false, false);
        c[3] = __builtin_amdgcn_wmma_f32_16x16x32_bf16(false, aC, false, bC3,
                                                       (short)0, c[3], false, false);
        aC = aN; bC0 = bN0; bC1 = bN1; bC2 = bN2; bC3 = bN3;
    }
    c[0] = __builtin_amdgcn_wmma_f32_16x16x32_bf16(false, aC, false, bC0,
                                                   (short)0, c[0], false, false);
    c[1] = __builtin_amdgcn_wmma_f32_16x16x32_bf16(false, aC, false, bC1,
                                                   (short)0, c[1], false, false);
    c[2] = __builtin_amdgcn_wmma_f32_16x16x32_bf16(false, aC, false, bC2,
                                                   (short)0, c[2], false, false);
    c[3] = __builtin_amdgcn_wmma_f32_16x16x32_bf16(false, aC, false, bC3,
                                                   (short)0, c[3], false, false);

    const int rm = (lane >> 4) * 8;       // C/D row offset for this half-wave
#pragma unroll
    for (int j = 0; j < 4; ++j) {
        const int hj = h0 + 16 * j + ncol;
        const float bias = hb[hj];
#pragma unroll
        for (int r = 0; r < 8; ++r) {
            const int bi = b0 + r + rm;
            float pre = c[j][r] + bias;
            float p = sigmoidf(pre);
            if (mode == 0) {
                float u = rnd01((uint32_t)(bi * HSZ + hj), seed);
                hout[(size_t)bi * HSZ + hj] = (bf16)((u < p) ? 1.0f : 0.0f);
            } else {
                phout[(size_t)bi * HSZ + hj] = p;
                if (mode == 2) {
                    float sp = (pre > 20.0f) ? pre : log1pf(__expf(pre));
                    atomicAdd(&logp[bi], sp);
                }
            }
        }
    }
}

// ---------------------------------------------------------------------------
// Visible pass: pv[b,v] = sigmoid(sum_h h[b,h]*W[h,v] + vb[v]); sample to vout
// grid = (B/16, V/64), block = 32: one wave computes a 16x64 strip.
// ---------------------------------------------------------------------------
__global__ __launch_bounds__(32)
void visible_wmma_kernel(const bf16* __restrict__ hin,
                         const bf16* __restrict__ WTbf,
                         const float* __restrict__ vb,
                         bf16* __restrict__ vout,
                         uint32_t seed) {
    const int b0   = blockIdx.x * 16;
    const int v0   = blockIdx.y * 64;
    const int lane = threadIdx.x;
    const int mrow = lane & 15;
    const int koff = (lane >> 4) * 16;
    const int ncol = lane & 15;
    const int khalf = (lane >> 4) * 16;

    v8f c[4] = {};
    const bf16* arow = hin + (size_t)(b0 + mrow) * HSZ + koff;
    const bf16* brow[4];
#pragma unroll
    for (int j = 0; j < 4; ++j)
        brow[j] = WTbf + (size_t)(v0 + 16 * j + ncol) * HSZ + khalf;

    v16bf aC  = make_a(arow);
    v16bf bC0 = *(const v16bf*)(brow[0]);
    v16bf bC1 = *(const v16bf*)(brow[1]);
    v16bf bC2 = *(const v16bf*)(brow[2]);
    v16bf bC3 = *(const v16bf*)(brow[3]);

#pragma unroll 2
    for (int k0 = 0; k0 < HSZ - 32; k0 += 32) {
        const int k1 = k0 + 32;
        v16bf aN  = make_a(arow + k1);
        v16bf bN0 = *(const v16bf*)(brow[0] + k1);
        v16bf bN1 = *(const v16bf*)(brow[1] + k1);
        v16bf bN2 = *(const v16bf*)(brow[2] + k1);
        v16bf bN3 = *(const v16bf*)(brow[3] + k1);
        c[0] = __builtin_amdgcn_wmma_f32_16x16x32_bf16(false, aC, false, bC0,
                                                       (short)0, c[0], false, false);
        c[1] = __builtin_amdgcn_wmma_f32_16x16x32_bf16(false, aC, false, bC1,
                                                       (short)0, c[1], false, false);
        c[2] = __builtin_amdgcn_wmma_f32_16x16x32_bf16(false, aC, false, bC2,
                                                       (short)0, c[2], false, false);
        c[3] = __builtin_amdgcn_wmma_f32_16x16x32_bf16(false, aC, false, bC3,
                                                       (short)0, c[3], false, false);
        aC = aN; bC0 = bN0; bC1 = bN1; bC2 = bN2; bC3 = bN3;
    }
    c[0] = __builtin_amdgcn_wmma_f32_16x16x32_bf16(false, aC, false, bC0,
                                                   (short)0, c[0], false, false);
    c[1] = __builtin_amdgcn_wmma_f32_16x16x32_bf16(false, aC, false, bC1,
                                                   (short)0, c[1], false, false);
    c[2] = __builtin_amdgcn_wmma_f32_16x16x32_bf16(false, aC, false, bC2,
                                                   (short)0, c[2], false, false);
    c[3] = __builtin_amdgcn_wmma_f32_16x16x32_bf16(false, aC, false, bC3,
                                                   (short)0, c[3], false, false);

    const int rm = (lane >> 4) * 8;
#pragma unroll
    for (int j = 0; j < 4; ++j) {
        const int vj = v0 + 16 * j + ncol;
        const float bias = vb[vj];
#pragma unroll
        for (int r = 0; r < 8; ++r) {
            const int bi = b0 + r + rm;
            float p = sigmoidf(c[j][r] + bias);
            float u = rnd01((uint32_t)(bi * VSZ + vj), seed);
            vout[(size_t)bi * VSZ + vj] = (bf16)((u < p) ? 1.0f : 0.0f);
        }
    }
}

// ---------------------------------------------------------------------------
// inv[b] = exp(-(logp[b] + dot(vk_b, vb))) / B
// ---------------------------------------------------------------------------
__global__ void inv_norm_kernel(const float* __restrict__ logp,
                                const bf16* __restrict__ vk,
                                const float* __restrict__ vb,
                                float* __restrict__ inv) {
    int b = blockIdx.x * blockDim.x + threadIdx.x;
    if (b >= BSZ) return;
    const bf16* row = vk + (size_t)b * VSZ;
    float dot = 0.0f;
    for (int v = 0; v < VSZ; ++v) dot += (float)row[v] * vb[v];
    inv[b] = __expf(-(logp[b] + dot)) * (1.0f / (float)BSZ);
}

// ---------------------------------------------------------------------------
// phT[h*B + b] = bf16(inv[b] * ph[b*H + h])   (transpose + per-sample scale)
// ---------------------------------------------------------------------------
__global__ void tscale_kernel(const float* __restrict__ ph,
                              const float* __restrict__ inv,
                              bf16* __restrict__ phT) {
    size_t idx = (size_t)blockIdx.x * blockDim.x + threadIdx.x;
    if (idx >= (size_t)BSZ * HSZ) return;
    int b = (int)(idx / HSZ);
    int h = (int)(idx % HSZ);
    phT[(size_t)h * BSZ + b] = (bf16)(ph[idx] * inv[b]);
}

// ---------------------------------------------------------------------------
// g_W[h,v] = sum_b phkT_s[h,b]*vk[b,v] - sum_b ph0T_s[h,b]*v0[b,v]
// Two WMMA chains per wave, K = B = 16384.  B-tiles staged through LDS with an
// in-LDS transpose so each lane reads a contiguous 32B column vector.
// grid = (H/16, V/16), block = 32
// ---------------------------------------------------------------------------
__global__ __launch_bounds__(32)
void gradw_wmma_kernel(const bf16* __restrict__ phkT,
                       const bf16* __restrict__ ph0T,
                       const bf16* __restrict__ vkbf,
                       const bf16* __restrict__ v0bf,
                       float* __restrict__ gw) {
    const int h0   = blockIdx.x * 16;
    const int v0   = blockIdx.y * 16;
    const int lane = threadIdx.x;
    const int mrow = lane & 15;
    const int koff = (lane >> 4) * 16;
    const int ncol = lane & 15;
    const int khalf = (lane >> 4) * 16;

    __shared__ __align__(32) bf16 sVk[16 * 32];   // [n][k]
    __shared__ __align__(32) bf16 sV0[16 * 32];

    v8f c1 = {}, c2 = {};
    const bf16* a1row = phkT + (size_t)(h0 + mrow) * BSZ + koff;
    const bf16* a2row = ph0T + (size_t)(h0 + mrow) * BSZ + koff;

    for (int kb = 0; kb < BSZ; kb += 32) {
        v16bf a1 = make_a(a1row + kb);
        v16bf a2 = make_a(a2row + kb);

        // cooperative load of 32 rows x 16 cols of each B matrix, transposed
        {
            const int t = lane;                       // row b = kb + t
            v16bf rk = *(const v16bf*)(vkbf + (size_t)(kb + t) * VSZ + v0);
            v16bf r0 = *(const v16bf*)(v0bf + (size_t)(kb + t) * VSZ + v0);
#pragma unroll
            for (int n = 0; n < 16; ++n) {
                sVk[n * 32 + t] = rk[n];
                sV0[n * 32 + t] = r0[n];
            }
        }
        __syncthreads();

        v16bf b1 = *(const v16bf*)&sVk[ncol * 32 + khalf];
        v16bf b2 = *(const v16bf*)&sV0[ncol * 32 + khalf];
        c1 = __builtin_amdgcn_wmma_f32_16x16x32_bf16(false, a1, false, b1,
                                                     (short)0, c1, false, false);
        c2 = __builtin_amdgcn_wmma_f32_16x16x32_bf16(false, a2, false, b2,
                                                     (short)0, c2, false, false);
        __syncthreads();
    }

    const int rm = (lane >> 4) * 8;
#pragma unroll
    for (int r = 0; r < 8; ++r) {
        int h = h0 + r + rm;
        int v = v0 + ncol;
        gw[(size_t)h * VSZ + v] = c1[r] - c2[r];
    }
}

// ---------------------------------------------------------------------------
// g_vb[v] = sum_b inv[b]*(vk[b,v] - batch[b,v])
// ---------------------------------------------------------------------------
__global__ void gvb_kernel(const float* __restrict__ inv,
                           const bf16* __restrict__ vkbf,
                           const float* __restrict__ batch,
                           float* __restrict__ gvb) {
    int v = blockIdx.x * blockDim.x + threadIdx.x;
    if (v >= VSZ) return;
    float acc = 0.0f;
    for (int b = 0; b < BSZ; ++b) {
        size_t idx = (size_t)b * VSZ + v;
        acc += inv[b] * ((float)vkbf[idx] - batch[idx]);
    }
    gvb[v] = acc;
}

// ---------------------------------------------------------------------------
// g_hb[h] = sum_b inv[b]*(phk[b,h] - ph0[b,h])
// ---------------------------------------------------------------------------
__global__ void ghb_kernel(const float* __restrict__ inv,
                           const float* __restrict__ phk,
                           const float* __restrict__ ph0,
                           float* __restrict__ ghb) {
    int h = blockIdx.x * blockDim.x + threadIdx.x;
    if (h >= HSZ) return;
    float acc = 0.0f;
    for (int b = 0; b < BSZ; ++b) {
        size_t idx = (size_t)b * HSZ + h;
        acc += inv[b] * (phk[idx] - ph0[idx]);
    }
    ghb[h] = acc;
}

// ---------------------------------------------------------------------------
static inline size_t al256(size_t x) { return (x + 255) & ~(size_t)255; }

extern "C" void kernel_launch(void* const* d_in, const int* in_sizes, int n_in,
                              void* d_out, int out_size, void* d_ws, size_t ws_size,
                              hipStream_t stream) {
    const float* batch = (const float*)d_in[0];   // [B,V]
    const float* W     = (const float*)d_in[1];   // [H,V]
    const float* vb    = (const float*)d_in[2];   // [V]
    const float* hb    = (const float*)d_in[3];   // [H]
    // d_in[4] is k (==8): device scalar, hard-coded as KGIBBS (graph-capture safe)

    float* out  = (float*)d_out;
    float* gw   = out;                        // H*V
    float* gvb  = out + (size_t)HSZ * VSZ;    // V
    float* ghb  = gvb + VSZ;                  // H

    // ---- workspace carve-up ------------------------------------------------
    char* ws = (char*)d_ws;
    size_t off = 0;
    bf16*  Wbf      = (bf16*)(ws + off); off = al256(off + (size_t)HSZ * VSZ * 2);
    bf16*  WTbf     = (bf16*)(ws + off); off = al256(off + (size_t)VSZ * HSZ * 2);
    bf16*  batch_bf = (bf16*)(ws + off); off = al256(off + (size_t)BSZ * VSZ * 2);
    bf16*  vbuf     = (bf16*)(ws + off); off = al256(off + (size_t)BSZ * VSZ * 2);
    bf16*  hbuf     = (bf16*)(ws + off); off = al256(off + (size_t)BSZ * HSZ * 2);
    float* ph0      = (float*)(ws + off); off = al256(off + (size_t)BSZ * HSZ * 4);
    float* phk      = (float*)(ws + off); off = al256(off + (size_t)BSZ * HSZ * 4);
    float* logp     = (float*)(ws + off); off = al256(off + (size_t)BSZ * 4);
    float* inv      = (float*)(ws + off); off = al256(off + (size_t)BSZ * 4);
    bf16*  phkT     = (bf16*)(ws + off); off = al256(off + (size_t)HSZ * BSZ * 2);
    bf16*  ph0T     = (bf16*)(ws + off); off = al256(off + (size_t)HSZ * BSZ * 2);
    (void)ws_size; (void)n_in; (void)in_sizes; (void)out_size;

    // ---- pre-conversion ----------------------------------------------------
    convert_w_kernel<<<(HSZ * VSZ + 255) / 256, 256, 0, stream>>>(W, Wbf, WTbf);
    convert_batch_kernel<<<((size_t)BSZ * VSZ + 255) / 256, 256, 0, stream>>>(
        batch, batch_bf, vbuf);
    zero_f32_kernel<<<(BSZ + 255) / 256, 256, 0, stream>>>(logp, BSZ);

    // ---- k steps of block Gibbs sampling (WMMA GEMMs) ----------------------
    const dim3 gridH(BSZ / 16, HSZ / 64);   // 1024 x 4
    const dim3 gridV(BSZ / 16, VSZ / 64);   // 1024 x 16
    for (int s = 0; s < KGIBBS; ++s) {
        hidden_wmma_kernel<<<gridH, 32, 0, stream>>>(
            vbuf, Wbf, hb, hbuf, nullptr, nullptr, /*mode=*/0,
            0xC0FFEE00u + (uint32_t)(2 * s));
        visible_wmma_kernel<<<gridV, 32, 0, stream>>>(
            hbuf, WTbf, vb, vbuf, 0xC0FFEE00u + (uint32_t)(2 * s + 1));
    }

    // ---- positive / negative hidden probabilities --------------------------
    hidden_wmma_kernel<<<gridH, 32, 0, stream>>>(
        batch_bf, Wbf, hb, nullptr, ph0, nullptr, /*mode=*/1, 0u);
    hidden_wmma_kernel<<<gridH, 32, 0, stream>>>(
        vbuf, Wbf, hb, nullptr, phk, logp, /*mode=*/2, 0u);

    // ---- inverse norm weights ----------------------------------------------
    inv_norm_kernel<<<(BSZ + 255) / 256, 256, 0, stream>>>(logp, vbuf, vb, inv);

    // ---- scaled transposes for the gradient GEMM ---------------------------
    tscale_kernel<<<((size_t)BSZ * HSZ + 255) / 256, 256, 0, stream>>>(phk, inv, phkT);
    tscale_kernel<<<((size_t)BSZ * HSZ + 255) / 256, 256, 0, stream>>>(ph0, inv, ph0T);

    // ---- g_W: two WMMA chains over K = 16384 -------------------------------
    const dim3 gridG(HSZ / 16, VSZ / 16);   // 16 x 64
    gradw_wmma_kernel<<<gridG, 32, 0, stream>>>(phkT, ph0T, vbuf, batch_bf, gw);

    // ---- bias gradients ----------------------------------------------------
    gvb_kernel<<<(VSZ + 255) / 256, 256, 0, stream>>>(inv, vbuf, batch, gvb);
    ghb_kernel<<<(HSZ + 255) / 256, 256, 0, stream>>>(inv, phk, ph0, ghb);
}
